// GraphormerAttentionHead_20985210208827
// MI455X (gfx1250) — matrix operand order repaired and checked by reading, not compiled, not run on hardware
//
#include <hip/hip_runtime.h>
#include <cstddef>
#include <cstdint>

// ---------------------------------------------------------------------------
// GraphormerAttentionHead on MI455X (gfx1250, wave32, WMMA)
//   q,k,v = x@W + b  (8192x256 -> 8192x64), bf16 WMMA w/ f32 accum
//   flash-style masked softmax attention w/ 8MB adjacency BITMASK and
//   double-buffered async global->LDS staging of K/V chunks (ASYNCcnt).
//   V kept TRANSPOSED in global so PV B-fragments are contiguous b128 reads;
//   LDS tiles padded to conflict-free row strides.
// ---------------------------------------------------------------------------

#define N_NODES 8192
#define DIM_IN  256
#define DIM_D   64
#define MASK_WORDS_PER_ROW (N_NODES / 32)   // 256 u32 per row

#define KPAD 72   // kbuf row stride (elems): 144B, 16B-aligned, conflict-free
#define VPAD 40   // vbufT row stride (elems): 80B, 16B-aligned, conflict-free

typedef __bf16 v16bf __attribute__((ext_vector_type(16)));
typedef __bf16 bf8v  __attribute__((ext_vector_type(8)));
typedef __bf16 bf2   __attribute__((ext_vector_type(2)));
typedef float  v8f   __attribute__((ext_vector_type(8)));
typedef float  f2    __attribute__((ext_vector_type(2)));

__device__ __forceinline__ v8f wmma_bf16(v16bf a, v16bf b, v8f c) {
  // v_wmma_f32_16x16x32_bf16: D = A(16x32) * B(32x16) + C, f32 accumulate
  return __builtin_amdgcn_wmma_f32_16x16x32_bf16(
      /*neg_a=*/false, a, /*neg_b=*/false, b,
      /*c_mod=*/(short)0, c, /*reuse_a=*/false, /*reuse_b=*/false);
}

// A-matrix (16x32, 16-bit) per-lane K index for element j (ISA 7.12.2):
//  lanes 0-15 : v0-3 -> K 0..7,  v4-7 -> K 16..23
//  lanes 16-31: v0-3 -> K 8..15, v4-7 -> K 24..31
__device__ __forceinline__ int akoff(int j, int half) {
  return (j < 8) ? (8 * half + j) : (16 + 8 * half + (j - 8));
}

// ---------------------------------------------------------------------------
// Kernel 1: zero the adjacency bitmask (8192*256 u32 = 8MB) as uint4 stores
// ---------------------------------------------------------------------------
__global__ void __launch_bounds__(256) zero_mask_kernel(uint4* __restrict__ m) {
  size_t i = (size_t)blockIdx.x * blockDim.x + threadIdx.x;
  uint4 z; z.x = 0u; z.y = 0u; z.z = 0u; z.w = 0u;
  m[i] = z;
}

// ---------------------------------------------------------------------------
// Kernel 2: scatter edges into the bitmask: adj[row, col] = 1
// edge_index flat layout: [0..E) = rows (edge_index[0]), [E..2E) = cols
// ---------------------------------------------------------------------------
__global__ void __launch_bounds__(256) scatter_adj_kernel(
    const int* __restrict__ ei, int E, unsigned* __restrict__ adj) {
  int t = blockIdx.x * blockDim.x + threadIdx.x;
  if (t >= E) return;
  int r = ei[t];
  int c = ei[E + t];
  atomicOr(&adj[(size_t)r * MASK_WORDS_PER_ROW + (c >> 5)], 1u << (c & 31));
}

// ---------------------------------------------------------------------------
// Kernel 3: projections q/k/v = x@W + b, written as bf16.
// Grid: 512 blocks x 128 threads. Block = 16 rows; wave w = 16-col tile w.
// Q,K stored row-major [node][d]; V stored TRANSPOSED [d][node].
// ---------------------------------------------------------------------------
__device__ __forceinline__ v8f proj_acc(const v16bf* __restrict__ a,
                                        const float* __restrict__ W,
                                        int colBase, int lr, int half) {
  v8f acc = {};
  #pragma unroll
  for (int ch = 0; ch < 8; ++ch) {           // 256 = 8 chunks of K=32
    v16bf bfrag;                             // B(32x16): lane=col, elem j: K=16*half+j
    #pragma unroll
    for (int j = 0; j < 16; ++j)
      bfrag[j] = (__bf16)W[(size_t)(ch * 32 + 16 * half + j) * DIM_D + colBase + lr];
    acc = wmma_bf16(a[ch], bfrag, acc);
  }
  return acc;
}

__global__ void __launch_bounds__(128) proj_kernel(
    const float* __restrict__ x,
    const float* __restrict__ Wq, const float* __restrict__ bq,
    const float* __restrict__ Wk, const float* __restrict__ bk,
    const float* __restrict__ Wv, const float* __restrict__ bv,
    __bf16* __restrict__ Qb, __bf16* __restrict__ Kb, __bf16* __restrict__ Vt) {
  const int lane = threadIdx.x & 31;
  const int wave = threadIdx.x >> 5;     // 0..3 -> output col tile
  const int half = lane >> 4;
  const int lr   = lane & 15;
  const int rowBase = blockIdx.x * 16;
  const int colBase = wave * 16;

  // A fragments: x rows in bf16, all 8 K-chunks, reused for q/k/v
  v16bf a[8];
  const float* xrow = x + (size_t)(rowBase + lr) * DIM_IN;
  #pragma unroll
  for (int ch = 0; ch < 8; ++ch) {
    #pragma unroll
    for (int i = 0; i < 8; ++i) {
      int k0 = ch * 32 + 16 * half + 2 * i;
      f2 p = *reinterpret_cast<const f2*>(xrow + k0);
      a[ch][2 * i]     = (__bf16)p[0];
      a[ch][2 * i + 1] = (__bf16)p[1];
    }
  }

  // Q: row-major
  {
    v8f acc = proj_acc(a, Wq, colBase, lr, half);
    float b = bq[colBase + lr];
    #pragma unroll
    for (int j = 0; j < 8; ++j)            // C elem j -> row 8*half+j, col lr
      Qb[(size_t)(rowBase + 8 * half + j) * DIM_D + colBase + lr] =
          (__bf16)(acc[j] + b);
  }
  // K: row-major
  {
    v8f acc = proj_acc(a, Wk, colBase, lr, half);
    float b = bk[colBase + lr];
    #pragma unroll
    for (int j = 0; j < 8; ++j)
      Kb[(size_t)(rowBase + 8 * half + j) * DIM_D + colBase + lr] =
          (__bf16)(acc[j] + b);
  }
  // V: transposed  Vt[d][node]
  {
    v8f acc = proj_acc(a, Wv, colBase, lr, half);
    float b = bv[colBase + lr];
    #pragma unroll
    for (int j = 0; j < 8; ++j)
      Vt[(size_t)(colBase + lr) * N_NODES + rowBase + 8 * half + j] =
          (__bf16)(acc[j] + b);
  }
}

// ---------------------------------------------------------------------------
// Kernel 4: flash-style masked attention with async K/V staging.
// Grid: 128 blocks x 128 threads; wave w owns 16 query rows.
// Per 32-key chunk: async-DMA next K/V chunk to LDS (double buffered),
// scores (4 WMMA), mask + online softmax, P transpose via LDS, PV (4 WMMA).
// ---------------------------------------------------------------------------
__global__ void __launch_bounds__(128) attn_kernel(
    const __bf16* __restrict__ Qb, const __bf16* __restrict__ Kb,
    const __bf16* __restrict__ Vt, const unsigned* __restrict__ adj,
    float* __restrict__ out) {
  __shared__ __bf16 kbuf[2][32][KPAD];   // K chunk, rows = keys   (padded)
  __shared__ __bf16 vbufT[2][DIM_D][VPAD]; // V chunk, rows = d-col (padded)
  __shared__ __bf16 plds[4][16][32];     // per-wave P transpose staging (4KB)

  const int tid  = threadIdx.x;
  const int lane = tid & 31;
  const int wave = tid >> 5;
  const int half = lane >> 4;
  const int lr   = lane & 15;
  const int qBase = blockIdx.x * 64 + wave * 16;

  // Async stage K chunk: 32 rows x 64 elems (128B = 8 x b128 per row).
  auto stageK = [&](const __bf16* __restrict__ g, __bf16* l) {
    unsigned lbase = (unsigned)(uintptr_t)l;     // low 32 bits = LDS offset
    #pragma unroll
    for (int i = 0; i < 2; ++i) {
      int seg = tid + i * 128;                   // row = seg>>3, part = seg&7
      unsigned loff = lbase + (unsigned)(seg >> 3) * (KPAD * 2) + (seg & 7) * 16;
      unsigned long long ga = (unsigned long long)(uintptr_t)
          (g + (size_t)(seg >> 3) * DIM_D + (seg & 7) * 8);
      asm volatile("global_load_async_to_lds_b128 %0, %1, off"
                   :: "v"(loff), "v"(ga) : "memory");
    }
  };
  // Async stage Vt chunk: 64 rows x 32 elems (64B = 4 x b128 per row).
  auto stageV = [&](const __bf16* __restrict__ g, __bf16* l) {
    unsigned lbase = (unsigned)(uintptr_t)l;
    #pragma unroll
    for (int i = 0; i < 2; ++i) {
      int seg = tid + i * 128;                   // row = seg>>2, part = seg&3
      unsigned loff = lbase + (unsigned)(seg >> 2) * (VPAD * 2) + (seg & 3) * 16;
      unsigned long long ga = (unsigned long long)(uintptr_t)
          (g + (size_t)(seg >> 2) * N_NODES + (seg & 3) * 8);
      asm volatile("global_load_async_to_lds_b128 %0, %1, off"
                   :: "v"(loff), "v"(ga) : "memory");
    }
  };

  // Q fragments (d=64 -> two K=32 chunks), loaded once
  v16bf qf[2];
  {
    const __bf16* qrow = Qb + (size_t)(qBase + lr) * DIM_D;
    #pragma unroll
    for (int ch = 0; ch < 2; ++ch) {
      #pragma unroll
      for (int i = 0; i < 8; ++i) {
        int k0 = ch * 32 + 16 * half + 2 * i;
        bf2 p = *reinterpret_cast<const bf2*>(qrow + k0);
        qf[ch][2 * i]     = p[0];
        qf[ch][2 * i + 1] = p[1];
      }
    }
  }

  float mrow[8], lrow[8];                // running max / sum, row = 8*half+j
  v8f acc[4];                            // 16x64 f32 output accumulator
  #pragma unroll
  for (int j = 0; j < 8; ++j) { mrow[j] = -__builtin_inff(); lrow[j] = 0.f; }
  v8f zero8 = {};
  #pragma unroll
  for (int t = 0; t < 4; ++t) acc[t] = zero8;

  const float scale = 0.125f;            // 1/sqrt(64)

  // Prologue: stage chunk 0 into buffer 0
  stageK(Kb, &kbuf[0][0][0]);
  stageV(Vt, &vbufT[0][0][0]);

  for (int kci = 0; kci < N_NODES / 32; ++kci) {
    const int kc  = kci * 32;
    const int buf = kci & 1;

    asm volatile("s_wait_asynccnt 0x0" ::: "memory");  // our DMA done
    __syncthreads();                                   // everyone's DMA + prior reads done

    if (kci + 1 < N_NODES / 32) {      // overlap next chunk's DMA with compute
      stageK(Kb + (size_t)(kc + 32) * DIM_D, &kbuf[buf ^ 1][0][0]);
      stageV(Vt + (kc + 32),                 &vbufT[buf ^ 1][0][0]);
    }

    // ---- scores: S(16q x 32k) as two 16x16 tiles, contraction over d=64 ----
    v8f s[2];
    #pragma unroll
    for (int t = 0; t < 2; ++t) {
      const __bf16* krow = &kbuf[buf][t * 16 + lr][0];  // lane = key col
      bf8v a0 = *reinterpret_cast<const bf8v*>(krow + 16 * half);       // d 16h..
      bf8v a1 = *reinterpret_cast<const bf8v*>(krow + 16 * half + 8);
      bf8v b0 = *reinterpret_cast<const bf8v*>(krow + 32 + 16 * half);  // d 32+16h..
      bf8v b1 = *reinterpret_cast<const bf8v*>(krow + 40 + 16 * half);
      v16bf kf0, kf1;
      #pragma unroll
      for (int j = 0; j < 8; ++j) {
        kf0[j] = a0[j]; kf0[8 + j] = a1[j];
        kf1[j] = b0[j]; kf1[8 + j] = b1[j];
      }
      v8f c = zero8;
      c = wmma_bf16(qf[0], kf0, c);
      c = wmma_bf16(qf[1], kf1, c);
      s[t] = c;
    }

    // ---- mask (1 u32 word == this 32-key chunk) + scale ----
    unsigned mw[8];
    const int wordIdx = kc >> 5;
    #pragma unroll
    for (int j = 0; j < 8; ++j)
      mw[j] = adj[(size_t)(qBase + 8 * half + j) * MASK_WORDS_PER_ROW + wordIdx];

    float cmax[8];
    #pragma unroll
    for (int j = 0; j < 8; ++j) cmax[j] = -1.0e6f;
    #pragma unroll
    for (int t = 0; t < 2; ++t) {
      int bit = t * 16 + lr;
      #pragma unroll
      for (int j = 0; j < 8; ++j) {
        float v = s[t][j] * scale;
        v = ((mw[j] >> bit) & 1u) ? v : -1.0e6f;
        s[t][j] = v;
        cmax[j] = fmaxf(cmax[j], v);
      }
    }
    // row reduce across the 16-lane half-group (xor 1,2,4,8 stays in-group)
    #pragma unroll
    for (int off = 1; off < 16; off <<= 1)
      #pragma unroll
      for (int j = 0; j < 8; ++j)
        cmax[j] = fmaxf(cmax[j], __shfl_xor(cmax[j], off, 32));

    // ---- online softmax update ----
    float corr[8];
    #pragma unroll
    for (int j = 0; j < 8; ++j) {
      float mn = fmaxf(mrow[j], cmax[j]);
      corr[j]  = __expf(mrow[j] - mn);   // exp(-inf - finite) = 0 first chunk
      mrow[j]  = mn;
    }
    float rsum[8];
    #pragma unroll
    for (int j = 0; j < 8; ++j) {
      float p0 = __expf(s[0][j] - mrow[j]);
      float p1 = __expf(s[1][j] - mrow[j]);
      s[0][j] = p0; s[1][j] = p1;
      rsum[j] = p0 + p1;
    }
    #pragma unroll
    for (int off = 1; off < 16; off <<= 1)
      #pragma unroll
      for (int j = 0; j < 8; ++j)
        rsum[j] += __shfl_xor(rsum[j], off, 32);
    #pragma unroll
    for (int j = 0; j < 8; ++j)
      lrow[j] = lrow[j] * corr[j] + rsum[j];
    #pragma unroll
    for (int t = 0; t < 4; ++t)
      #pragma unroll
      for (int j = 0; j < 8; ++j)
        acc[t][j] *= corr[j];

    // ---- P (C layout) -> LDS -> A-fragment layout (cross-lane transpose) ----
    #pragma unroll
    for (int t = 0; t < 2; ++t)
      #pragma unroll
      for (int j = 0; j < 8; ++j)
        plds[wave][8 * half + j][t * 16 + lr] = (__bf16)s[t][j];
    asm volatile("s_wait_dscnt 0" ::: "memory");   // intra-wave LDS RAW
    v16bf pf;
    #pragma unroll
    for (int i = 0; i < 8; ++i) {
      int k0 = 16 * half + 2 * i;
      bf2 p = *reinterpret_cast<const bf2*>(&plds[wave][lr][k0]);
      pf[2 * i]     = p[0];
      pf[2 * i + 1] = p[1];
    }

    // ---- PV: P(16x32) @ V(32x64) as 4 column tiles (contiguous Vt reads) ----
    #pragma unroll
    for (int t4 = 0; t4 < 4; ++t4) {
      const __bf16* vrow = &vbufT[buf][t4 * 16 + lr][0]; // lane = out col
      bf8v v0 = *reinterpret_cast<const bf8v*>(vrow + 16 * half);      // keys 16h..
      bf8v v1 = *reinterpret_cast<const bf8v*>(vrow + 16 * half + 8);
      v16bf vf;                          // B elem j: key-in-chunk = 16*half + j
      #pragma unroll
      for (int j = 0; j < 8; ++j) { vf[j] = v0[j]; vf[8 + j] = v1[j]; }
      acc[t4] = wmma_bf16(pf, vf, acc[t4]);
    }
  }

  // ---- epilogue: out = acc / l ----
  #pragma unroll
  for (int t = 0; t < 4; ++t)
    #pragma unroll
    for (int j = 0; j < 8; ++j) {
      int r = qBase + 8 * half + j;
      out[(size_t)r * DIM_D + t * 16 + lr] = acc[t][j] / lrow[j];
    }
}

// ---------------------------------------------------------------------------
// Launch
// ---------------------------------------------------------------------------
extern "C" void kernel_launch(void* const* d_in, const int* in_sizes, int n_in,
                              void* d_out, int out_size, void* d_ws, size_t ws_size,
                              hipStream_t stream) {
  const float* x  = (const float*)d_in[0];
  const int*   ei = (const int*)  d_in[1];
  const float* Wq = (const float*)d_in[2];
  const float* bq = (const float*)d_in[3];
  const float* Wk = (const float*)d_in[4];
  const float* bk = (const float*)d_in[5];
  const float* Wv = (const float*)d_in[6];
  const float* bv = (const float*)d_in[7];
  float* out = (float*)d_out;

  const int E = in_sizes[1] / 2;

  // workspace: [0,8MB) adjacency bitmask; then bf16 Q, K (row-major) and
  // Vt (transposed [d][node]), 1MB each
  unsigned char* ws = (unsigned char*)d_ws;
  unsigned* mask = (unsigned*)ws;
  __bf16* Qb = (__bf16*)(ws + (size_t) 8 * 1024 * 1024);
  __bf16* Kb = (__bf16*)(ws + (size_t) 9 * 1024 * 1024);
  __bf16* Vt = (__bf16*)(ws + (size_t)10 * 1024 * 1024);

  // 8MB mask = 524288 uint4
  zero_mask_kernel<<<2048, 256, 0, stream>>>((uint4*)mask);
  scatter_adj_kernel<<<(E + 255) / 256, 256, 0, stream>>>(ei, E, mask);
  proj_kernel<<<N_NODES / 16, 128, 0, stream>>>(x, Wq, bq, Wk, bk, Wv, bv,
                                                Qb, Kb, Vt);
  attn_kernel<<<N_NODES / 64, 128, 0, stream>>>(Qb, Kb, Vt, mask, out);
}
